// TreeLSTMCell_5557687681541
// MI455X (gfx1250) — matrix operand order repaired and checked by reading, not compiled, not run on hardware
//
#include <hip/hip_runtime.h>

#define NND   32768
#define NCH   8
#define HH    256
#define PITCH 264   // bf16 elements per staged LDS row (256 + 8 pad vs bank conflicts)

typedef __attribute__((ext_vector_type(16))) __bf16 v16bf;
typedef __attribute__((ext_vector_type(8)))  __bf16 v8bf;
typedef __attribute__((ext_vector_type(8)))  float  v8f;
typedef __attribute__((ext_vector_type(4)))  float  f4v;

#define WMMA_BF16(A, B, C) \
  __builtin_amdgcn_wmma_f32_16x16x32_bf16(false, (A), false, (B), (short)0, (C), false, false)

__device__ __forceinline__ v8f zero8() {
  v8f v;
#pragma unroll
  for (int i = 0; i < 8; ++i) v[i] = 0.0f;
  return v;
}

// A fragment (16x32 bf16) from row-major bf16 LDS tile.
// ISA 7.12.2: lanes 0-15 (row M=lane) hold K = kb..kb+7 and kb+16..kb+23 with kb=0;
// lanes 16-31 hold the same with kb=8.
__device__ __forceinline__ v16bf lds_a_frag(const __bf16* base, int lane, int kt) {
  const int row = lane & 15;
  const int kb  = kt * 32 + ((lane >> 4) << 3);
  v8bf lo = *(const v8bf*)(base + row * PITCH + kb);
  v8bf hi = *(const v8bf*)(base + row * PITCH + kb + 16);
  return __builtin_shufflevector(lo, hi, 0,1,2,3,4,5,6,7,8,9,10,11,12,13,14,15);
}

// B fragment (32x16 bf16), pre-packed: 1024 B per fragment, 32 B contiguous per lane.
// These hit L2 (weights ~1 MB, resident) -> regular temporal policy.
__device__ __forceinline__ v16bf ld_b_frag(const unsigned short* frags, int nt, int kt, int lane) {
  return *(const v16bf*)(frags + (size_t)(nt * 8 + kt) * 512 + (size_t)lane * 16);
}

__device__ __forceinline__ float sigmoidf_(float v) { return 1.0f / (1.0f + __expf(-v)); }
__device__ __forceinline__ float tanhf_(float v)    { float e = __expf(2.0f * v); return (e - 1.0f) / (e + 1.0f); }

// ---------------------------------------------------------------------------
// Pack a row-major fp32 weight [K=256, ncols] into bf16 WMMA-B fragments.
// Fragment (nt,kt): lane l holds column nt*16+(l&15), K = kt*32+(j>>3)*16+(l>>4)*8+(j&7).
// ---------------------------------------------------------------------------
__global__ void __launch_bounds__(256) pack_b_kernel(const float* __restrict__ W,
                                                     unsigned short* __restrict__ dst,
                                                     int ncols) {
  const int frag = blockIdx.x;   // nt*8 + kt
  const int nt = frag >> 3;
  const int kt = frag & 7;
  for (int e = threadIdx.x; e < 512; e += 256) {
    const int lane = e >> 4;
    const int j    = e & 15;
    const int col  = nt * 16 + (lane & 15);
    const int k    = kt * 32 + ((j >> 3) << 4) + ((lane >> 4) << 3) + (j & 7);
    __bf16 b = (__bf16)W[(size_t)k * ncols + col];
    dst[(size_t)frag * 512 + e] = __builtin_bit_cast(unsigned short, b);
  }
}

// ---------------------------------------------------------------------------
// Fused TreeLSTM cell. One workgroup (8 waves) per 16-node tile.
// Wave w owns output columns [w*32, w*32+31] in H-space for f/c_aggr AND in
// each gate slice (i: n, o: n+256, u: n+512) -> gate combine stays in regs.
// n_h staging is double-buffered so the HBM stream for child c+1 overlaps the
// WMMA phase of child c. Streaming tensors use non-temporal policy so the
// L2-resident packed weights are not evicted.
// ---------------------------------------------------------------------------
__global__ void __launch_bounds__(256) tree_lstm_kernel(
    const float* __restrict__ x,
    const float* __restrict__ n_h,
    const float* __restrict__ n_c,
    const unsigned short* __restrict__ Bfin,   // W_f_in   packed (16 ntiles)
    const unsigned short* __restrict__ Bf,     // W_f      packed (16 ntiles)
    const unsigned short* __restrict__ Biou,   // W_iou_in packed (48 ntiles)
    const unsigned short* __restrict__ Bagg,   // W_aggr   packed (48 ntiles)
    const float* __restrict__ b_iou_in,
    const float* __restrict__ b_f_in,
    const float* __restrict__ b_f,
    const float* __restrict__ b_aggr,
    float* __restrict__ out) {
  __shared__ __align__(16) __bf16 xA[16 * PITCH];      // x tile, bf16 A staging
  __shared__ __align__(16) __bf16 nhA[2][16 * PITCH];  // double-buffered child n_h
  __shared__ __align__(16) __bf16 hsA[16 * PITCH];     // sum_c n_h tile

  const int tid  = threadIdx.x;
  const int wave = tid >> 5;
  const int lane = tid & 31;
  const int m0   = blockIdx.x << 4;

  const int row = tid >> 4;          // staging row 0..15
  const int cb  = (tid & 15) << 4;   // staging col base (16 floats per thread)

  float hs[16];
#pragma unroll
  for (int e = 0; e < 16; ++e) hs[e] = 0.0f;

  // ---- stage x tile + child 0 tile (fp32 global NT -> bf16 LDS) ----
  {
    const f4v* px = (const f4v*)(x + (size_t)(m0 + row) * HH + cb);
    const f4v* pn = (const f4v*)(n_h + ((size_t)(m0 + row) * NCH + 0) * HH + cb);
#pragma unroll
    for (int j = 0; j < 4; ++j) {
      f4v v = __builtin_nontemporal_load(px + j);
      f4v w = __builtin_nontemporal_load(pn + j);
#pragma unroll
      for (int q = 0; q < 4; ++q) {
        xA[row * PITCH + cb + j * 4 + q]     = (__bf16)v[q];
        hs[j * 4 + q]                       += w[q];
        nhA[0][row * PITCH + cb + j * 4 + q] = (__bf16)w[q];
      }
    }
  }
  __syncthreads();

  // ---- f_input = x @ W_f_in + b_f_in  (wave cols w*32 .. +31) ----
  v8f fi[2] = { zero8(), zero8() };
#pragma unroll
  for (int kt = 0; kt < 8; ++kt) {
    v16bf a = lds_a_frag(xA, lane, kt);
    fi[0] = WMMA_BF16(a, ld_b_frag(Bfin, wave * 2 + 0, kt, lane), fi[0]);
    fi[1] = WMMA_BF16(a, ld_b_frag(Bfin, wave * 2 + 1, kt, lane), fi[1]);
  }
  float bfv[2];
#pragma unroll
  for (int t = 0; t < 2; ++t) {
    const int n = wave * 32 + t * 16 + (lane & 15);
    const float bfin = b_f_in[n];
#pragma unroll
    for (int r = 0; r < 8; ++r) fi[t][r] += bfin;
    bfv[t] = b_f[n];
  }

  // ---- child loop: double-buffered; stage c+1 while computing c ----
  v8f ca[2] = { zero8(), zero8() };

  for (int c = 0; c < NCH; ++c) {
    const int cur = c & 1;
    // Stage next child into the other buffer (its prior readers finished at
    // the barrier that ended iteration c-1).
    if (c + 1 < NCH) {
      __bf16* dst = nhA[cur ^ 1];
      const f4v* pn =
          (const f4v*)(n_h + ((size_t)(m0 + row) * NCH + (c + 1)) * HH + cb);
#pragma unroll
      for (int j = 0; j < 4; ++j) {
        f4v w = __builtin_nontemporal_load(pn + j);
#pragma unroll
        for (int q = 0; q < 4; ++q) {
          hs[j * 4 + q]                 += w[q];
          dst[row * PITCH + cb + j * 4 + q] = (__bf16)w[q];
        }
      }
    }

    // f_child = n_h[:,c,:] @ W_f ; f = sigmoid(. + f_input + b_f); ca += f*n_c
    v8f fc[2] = { zero8(), zero8() };
#pragma unroll
    for (int kt = 0; kt < 8; ++kt) {
      v16bf a = lds_a_frag(nhA[cur], lane, kt);
      fc[0] = WMMA_BF16(a, ld_b_frag(Bf, wave * 2 + 0, kt, lane), fc[0]);
      fc[1] = WMMA_BF16(a, ld_b_frag(Bf, wave * 2 + 1, kt, lane), fc[1]);
    }
#pragma unroll
    for (int t = 0; t < 2; ++t) {
      const int n = wave * 32 + t * 16 + (lane & 15);
#pragma unroll
      for (int r = 0; r < 8; ++r) {
        const int m = ((lane >> 4) << 3) + r;   // C layout: M = r + 8*(lane>=16)
        const float fval = sigmoidf_(fc[t][r] + fi[t][r] + bfv[t]);
        const float nc = __builtin_nontemporal_load(
            n_c + ((size_t)(m0 + m) * NCH + c) * HH + n);
        ca[t][r] += fval * nc;
      }
    }
    __syncthreads();  // readers of nhA[cur] done; writes of nhA[cur^1] visible
  }

  // ---- stage h_sum as bf16 A tile ----
#pragma unroll
  for (int e = 0; e < 16; ++e) hsA[row * PITCH + cb + e] = (__bf16)hs[e];
  __syncthreads();

  // ---- iou = x @ W_iou_in + h_sum @ W_aggr (+ biases) ----
  v8f ai[2] = { zero8(), zero8() };
  v8f ao[2] = { zero8(), zero8() };
  v8f au[2] = { zero8(), zero8() };
#pragma unroll
  for (int kt = 0; kt < 8; ++kt) {
    v16bf a = lds_a_frag(xA, lane, kt);
#pragma unroll
    for (int t = 0; t < 2; ++t) {
      ai[t] = WMMA_BF16(a, ld_b_frag(Biou,      wave * 2 + t, kt, lane), ai[t]);
      ao[t] = WMMA_BF16(a, ld_b_frag(Biou, 16 + wave * 2 + t, kt, lane), ao[t]);
      au[t] = WMMA_BF16(a, ld_b_frag(Biou, 32 + wave * 2 + t, kt, lane), au[t]);
    }
  }
#pragma unroll
  for (int kt = 0; kt < 8; ++kt) {
    v16bf a = lds_a_frag(hsA, lane, kt);
#pragma unroll
    for (int t = 0; t < 2; ++t) {
      ai[t] = WMMA_BF16(a, ld_b_frag(Bagg,      wave * 2 + t, kt, lane), ai[t]);
      ao[t] = WMMA_BF16(a, ld_b_frag(Bagg, 16 + wave * 2 + t, kt, lane), ao[t]);
      au[t] = WMMA_BF16(a, ld_b_frag(Bagg, 32 + wave * 2 + t, kt, lane), au[t]);
    }
  }

  // ---- epilogue: gates, cell, hidden; non-temporal direct stores ----
#pragma unroll
  for (int t = 0; t < 2; ++t) {
    const int n = wave * 32 + t * 16 + (lane & 15);
    const float bi = b_iou_in[n      ] + b_aggr[n      ];
    const float bo = b_iou_in[n + 256] + b_aggr[n + 256];
    const float bu = b_iou_in[n + 512] + b_aggr[n + 512];
#pragma unroll
    for (int r = 0; r < 8; ++r) {
      const int m = m0 + ((lane >> 4) << 3) + r;
      const float iv = sigmoidf_(ai[t][r] + bi);
      const float uv = tanhf_(au[t][r] + bu);
      const float cv = iv * uv + ca[t][r];
      const float ov = sigmoidf_(ao[t][r] + bo);
      const float hv = ov * tanhf_(cv);
      __builtin_nontemporal_store(hv, out + (size_t)m * HH + n);                    // h
      __builtin_nontemporal_store(cv, out + (size_t)NND * HH + (size_t)m * HH + n); // c
    }
  }
}

extern "C" void kernel_launch(void* const* d_in, const int* in_sizes, int n_in,
                              void* d_out, int out_size, void* d_ws, size_t ws_size,
                              hipStream_t stream) {
  (void)in_sizes; (void)n_in; (void)out_size; (void)ws_size;

  const float* x        = (const float*)d_in[0];
  const float* n_h      = (const float*)d_in[1];
  const float* n_c      = (const float*)d_in[2];
  const float* W_iou_in = (const float*)d_in[3];
  const float* b_iou_in = (const float*)d_in[4];
  const float* W_f_in   = (const float*)d_in[5];
  const float* b_f_in   = (const float*)d_in[6];
  const float* W_f      = (const float*)d_in[7];
  const float* b_f      = (const float*)d_in[8];
  const float* W_aggr   = (const float*)d_in[9];
  const float* b_aggr   = (const float*)d_in[10];

  // Workspace: packed bf16 B-fragments (1024 frags x 1 KB = 1 MB total).
  unsigned short* ws   = (unsigned short*)d_ws;
  unsigned short* Bfin = ws;                     // 128 frags (16 nt x 8 kt)
  unsigned short* Bf   = ws + (size_t)128 * 512; // 128 frags
  unsigned short* Biou = ws + (size_t)256 * 512; // 384 frags (48 nt x 8 kt)
  unsigned short* Bagg = ws + (size_t)640 * 512; // 384 frags

  pack_b_kernel<<<128, 256, 0, stream>>>(W_f_in,   Bfin, 256);
  pack_b_kernel<<<128, 256, 0, stream>>>(W_f,      Bf,   256);
  pack_b_kernel<<<384, 256, 0, stream>>>(W_iou_in, Biou, 768);
  pack_b_kernel<<<384, 256, 0, stream>>>(W_aggr,   Bagg, 768);

  tree_lstm_kernel<<<NND / 16, 256, 0, stream>>>(
      x, n_h, n_c, Bfin, Bf, Biou, Bagg,
      b_iou_in, b_f_in, b_f, b_aggr, (float*)d_out);
}
